// NSAAttention_54812372632291
// MI455X (gfx1250) — matrix operand order repaired and checked by reading, not compiled, not run on hardware
//
#include <hip/hip_runtime.h>

// ---------------- constants ----------------
constexpr int kT   = 2048;
constexpr int kDim = 1024;
constexpr int kH   = 16;
constexpr int kD   = 64;
constexpr int kC   = 32;    // T / BS
constexpr int kBS  = 64;
constexpr int kW   = 256;
constexpr int kNQKV = 3 * kDim;       // 3072
#define SCALEQ 0.125f                  // D^-0.5

#define USE_ASYNC 1

typedef __attribute__((ext_vector_type(16))) __bf16 bf16x16;
typedef __attribute__((ext_vector_type(8)))  float  f32x8;

// load a 16-element bf16 fragment as two 16B chunks (per-lane)
__device__ __forceinline__ bf16x16 ld_frag(const __bf16* p0, const __bf16* p1) {
  union { bf16x16 v; uint4 q[2]; } u;
  u.q[0] = *(const uint4*)p0;
  u.q[1] = *(const uint4*)p1;
  return u.v;
}

__device__ __forceinline__ f32x8 wmma_bf16(bf16x16 a, bf16x16 b, f32x8 c) {
  return __builtin_amdgcn_wmma_f32_16x16x32_bf16(false, a, false, b, (short)0, c,
                                                 false, false);
}

// ---- CDNA5 async global->LDS copy (16B per lane), tracked by ASYNCcnt ----
__device__ __forceinline__ void async_ld16(__bf16* l, const __bf16* g) {
#if USE_ASYNC
  // LDS aperture keeps the LDS byte offset in the low 32 bits of a generic ptr
  unsigned lo = (unsigned)(unsigned long long)(const void*)l;
  asm volatile("global_load_async_to_lds_b128 %0, %1, off"
               :: "v"(lo), "v"(g) : "memory");
#else
  *(uint4*)l = *(const uint4*)g;
#endif
}

__device__ __forceinline__ void async_wait() {
#if USE_ASYNC
  asm volatile("s_wait_asynccnt 0x0" ::: "memory");
#endif
}

// ---------------- pack kernels ----------------
__global__ void pack_x_kernel(const float* __restrict__ x, __bf16* __restrict__ xb, int n) {
  int i = blockIdx.x * 256 + threadIdx.x;
  if (i < n) xb[i] = (__bf16)x[i];
}

// WqkvT[n][k] (bf16), n in [0,3072): q | k | v, transposed from [K][N] fp32
__global__ void pack_wqkv_kernel(const float* __restrict__ Wq, const float* __restrict__ Wk,
                                 const float* __restrict__ Wv, __bf16* __restrict__ WT) {
  int i = blockIdx.x * 256 + threadIdx.x;
  if (i >= kNQKV * kDim) return;
  int n = i >> 10, k = i & 1023;
  float v;
  if (n < kDim)            v = Wq[k * kDim + n];
  else if (n < 2 * kDim)   v = Wk[k * kDim + (n - kDim)];
  else                     v = Wv[k * kDim + (n - 2 * kDim)];
  WT[i] = (__bf16)v;
}

// WgT[n][k], n in [0,64), zero-padded past 48
__global__ void pack_wg_kernel(const float* __restrict__ Wg, __bf16* __restrict__ WT) {
  int i = blockIdx.x * 256 + threadIdx.x;
  if (i >= 64 * kDim) return;
  int n = i >> 10, k = i & 1023;
  WT[i] = (__bf16)((n < 48) ? Wg[k * 48 + n] : 0.f);
}

__global__ void pack_wo_kernel(const float* __restrict__ Wo, __bf16* __restrict__ WT) {
  int i = blockIdx.x * 256 + threadIdx.x;
  if (i >= kDim * kDim) return;
  int n = i >> 10, k = i & 1023;
  WT[i] = (__bf16)Wo[k * kDim + n];
}

// ---------------- generic bf16 GEMM: C[M][N] = A[M][K] * BT[N][K]^T ----------------
// block tile 128(M) x 64(N), 8 waves (2x4), k-chunk 32, double-buffered LDS with
// async global->LDS copies overlapping WMMA compute.
#define LSTR 40
__global__ __launch_bounds__(256) void gemm_bf16_kernel(
    const __bf16* __restrict__ A, const __bf16* __restrict__ BT,
    float* __restrict__ C, int M, int N, int K) {
  __shared__ __align__(16) __bf16 As[2][128 * LSTR];
  __shared__ __align__(16) __bf16 Bs[2][64 * LSTR];
  const int tid  = threadIdx.x;
  const int lane = tid & 31;
  const int wid  = tid >> 5;
  const int wm   = wid >> 2;           // 0..1  (64 rows each)
  const int wn   = wid & 3;            // 0..3  (16 cols each)
  const int n0   = blockIdx.x * 64;
  const int m0   = blockIdx.y * 128;
  const int ln   = lane & 15;
  const int half = lane >> 4;
  const int kOff = half << 3;          // 0 or 8

  // per-thread staging coordinates (fixed across iterations)
  const int ar0 = tid >> 2,          as0 = tid & 3;          // A vector 0
  const int ar1 = (tid + 256) >> 2,  as1 = (tid + 256) & 3;  // A vector 1
  const int br  = tid >> 2,          bs  = tid & 3;          // B vector

  f32x8 acc[4];
  for (int s = 0; s < 4; ++s) acc[s] = (f32x8)(0.0f);

  // prologue: stage k-tile 0 into buffer 0
  async_ld16(&As[0][ar0 * LSTR + as0 * 8], &A[(size_t)(m0 + ar0) * K + as0 * 8]);
  async_ld16(&As[0][ar1 * LSTR + as1 * 8], &A[(size_t)(m0 + ar1) * K + as1 * 8]);
  async_ld16(&Bs[0][br  * LSTR + bs  * 8], &BT[(size_t)(n0 + br) * K + bs * 8]);
  async_wait();
  __syncthreads();

  const int nk = K >> 5;
  for (int i = 0; i < nk; ++i) {
    const int p = i & 1;
    if (i + 1 < nk) {
      const int kk = (i + 1) << 5;
      async_ld16(&As[p ^ 1][ar0 * LSTR + as0 * 8],
                 &A[(size_t)(m0 + ar0) * K + kk + as0 * 8]);
      async_ld16(&As[p ^ 1][ar1 * LSTR + as1 * 8],
                 &A[(size_t)(m0 + ar1) * K + kk + as1 * 8]);
      async_ld16(&Bs[p ^ 1][br * LSTR + bs * 8],
                 &BT[(size_t)(n0 + br) * K + kk + bs * 8]);
    }
    // batch all fragment loads, then issue the 4 WMMAs
    bf16x16 bf = ld_frag(&Bs[p][(wn * 16 + ln) * LSTR + half * 8],
                         &Bs[p][(wn * 16 + ln) * LSTR + half * 8 + 16]);
    bf16x16 af[4];
    for (int s = 0; s < 4; ++s) {
      const __bf16* ap = &As[p][(wm * 64 + s * 16 + ln) * LSTR];
      af[s] = ld_frag(ap + kOff, ap + kOff + 16);
    }
    for (int s = 0; s < 4; ++s) acc[s] = wmma_bf16(af[s], bf, acc[s]);
    async_wait();
    __syncthreads();
  }
  const int cn = n0 + wn * 16 + ln;
  for (int s = 0; s < 4; ++s)
    for (int r = 0; r < 8; ++r) {
      int row = m0 + wm * 64 + s * 16 + r + 8 * half;
      C[(size_t)row * N + cn] = acc[s][r];
    }
}

// ---------------- RoPE + layout pack ----------------
// qkv_raw[t][3072] fp32 -> qb,kb [H][T][D] bf16 (roped), vT [H][D][T] bf16
__global__ void rope_pack_kernel(const float* __restrict__ qkv,
                                 const float* __restrict__ cosb,
                                 const float* __restrict__ sinb,
                                 __bf16* __restrict__ qb, __bf16* __restrict__ kb,
                                 __bf16* __restrict__ vT) {
  int id = blockIdx.x * 256 + threadIdx.x;
  if (id >= kT * kDim) return;
  int t = id >> 10;
  int hd = id & 1023;
  int h = hd >> 6, d = hd & 63;
  const float* row = qkv + (size_t)t * kNQKV;
  float q = row[hd], k = row[kDim + hd], v = row[2 * kDim + hd];
  int j = d & 31;
  float c = cosb[t * 32 + j], s = sinb[t * 32 + j];
  float qo, ko;
  if (d < 32) {
    qo = q * c - row[hd + 32] * s;
    ko = k * c - row[kDim + hd + 32] * s;
  } else {
    qo = q * c + row[hd - 32] * s;
    ko = k * c + row[kDim + hd - 32] * s;
  }
  size_t p = (size_t)h * kT * kD + (size_t)t * kD + d;
  qb[p] = (__bf16)qo;
  kb[p] = (__bf16)ko;
  vT[(size_t)h * kD * kT + (size_t)d * kT + t] = (__bf16)v;
}

// ---------------- block pooling (compression K/V) ----------------
// kcmp [H][C][D] bf16, vcmpT [H][D][C] bf16
__global__ void pool_cmp_kernel(const __bf16* __restrict__ kb, const __bf16* __restrict__ vT,
                                __bf16* __restrict__ kcmp, __bf16* __restrict__ vcmpT) {
  int b = blockIdx.x;              // H*C
  int h = b >> 5, c = b & 31;
  int d = threadIdx.x;             // 64
  float sk = 0.f, sv = 0.f;
  for (int r = 0; r < kBS; ++r) {
    sk += (float)kb[(size_t)h * kT * kD + (size_t)(c * kBS + r) * kD + d];
    sv += (float)vT[(size_t)h * kD * kT + (size_t)d * kT + c * kBS + r];
  }
  kcmp[(h * kC + c) * kD + d]  = (__bf16)(sk * (1.f / 64.f));
  vcmpT[(h * kD + d) * kC + c] = (__bf16)(sv * (1.f / 64.f));
}

// ---------------- compression attention ----------------
// one wave per (h, 16-row q tile). Writes w_cmp [H][T][C] f32 and o_cmp [H][T][D] f32.
__global__ __launch_bounds__(32) void cmp_attn_kernel(
    const __bf16* __restrict__ qb, const __bf16* __restrict__ kcmp,
    const __bf16* __restrict__ vcmpT, float* __restrict__ wcmp,
    float* __restrict__ ocmp) {
  __shared__ __align__(16) __bf16 P[16 * LSTR];
  const int h = blockIdx.x, t0 = blockIdx.y * 16;
  const int lane = threadIdx.x, half = lane >> 4, ln = lane & 15, kOff = half << 3;
  const float NEGINF = -__builtin_inff();

  const __bf16* qh = qb + (size_t)h * kT * kD;
  bf16x16 aq[2];
  for (int kc = 0; kc < 2; ++kc) {
    const __bf16* p = &qh[(size_t)(t0 + ln) * kD + kc * 32];
    aq[kc] = ld_frag(p + kOff, p + kOff + 16);
  }
  const __bf16* kch = kcmp + (size_t)h * kC * kD;
  f32x8 S[2];
  S[0] = (f32x8)(0.0f); S[1] = (f32x8)(0.0f);
  for (int nt = 0; nt < 2; ++nt)
    for (int kc = 0; kc < 2; ++kc) {
      const __bf16* p = &kch[(size_t)(nt * 16 + ln) * kD + kc * 32 + half * 8];
      S[nt] = wmma_bf16(aq[kc], ld_frag(p, p + 16), S[nt]);
    }
  // masked softmax over 32 blocks: valid iff t >= (c+1)*64
  float x[2][8], mr[8], rs[8];
  for (int r = 0; r < 8; ++r) mr[r] = NEGINF;
  for (int nt = 0; nt < 2; ++nt)
    for (int r = 0; r < 8; ++r) {
      int t = t0 + r + 8 * half, c = nt * 16 + ln;
      float v = S[nt][r] * SCALEQ;
      x[nt][r] = (t >= (c + 1) * kBS) ? v : NEGINF;
      mr[r] = fmaxf(mr[r], x[nt][r]);
    }
  for (int m = 1; m <= 8; m <<= 1)
    for (int r = 0; r < 8; ++r) mr[r] = fmaxf(mr[r], __shfl_xor(mr[r], m, 32));
  for (int r = 0; r < 8; ++r) rs[r] = 0.f;
  float e[2][8];
  for (int nt = 0; nt < 2; ++nt)
    for (int r = 0; r < 8; ++r) {
      e[nt][r] = (x[nt][r] == NEGINF) ? 0.f : __expf(x[nt][r] - mr[r]);
      rs[r] += e[nt][r];
    }
  for (int m = 1; m <= 8; m <<= 1)
    for (int r = 0; r < 8; ++r) rs[r] += __shfl_xor(rs[r], m, 32);
  for (int nt = 0; nt < 2; ++nt)
    for (int r = 0; r < 8; ++r) {
      int t = t0 + r + 8 * half, c = nt * 16 + ln;
      float w = e[nt][r] / fmaxf(rs[r], 1e-20f);
      wcmp[((size_t)h * kT + t) * kC + c] = w;
      P[(r + 8 * half) * LSTR + c] = (__bf16)w;
    }
  __syncthreads();
  bf16x16 ap = ld_frag(&P[ln * LSTR + kOff], &P[ln * LSTR + kOff + 16]);
  const __bf16* vch = vcmpT + (size_t)h * kD * kC;
  for (int n = 0; n < 4; ++n) {
    const __bf16* p = &vch[(size_t)(n * 16 + ln) * kC + half * 8];
    f32x8 o = wmma_bf16(ap, ld_frag(p, p + 16), (f32x8)(0.0f));
    for (int r = 0; r < 8; ++r) {
      int t = t0 + r + 8 * half;
      ocmp[((size_t)h * kT + t) * kD + n * 16 + ln] = o[r];
    }
  }
}

// ---------------- top-8 block selection ----------------
__global__ void select_topk_kernel(const float* __restrict__ wcmp,
                                   unsigned* __restrict__ selmask) {
  int id = blockIdx.x * 256 + threadIdx.x;       // H*T
  if (id >= kH * kT) return;
  int h = id >> 11, t = id & (kT - 1);
  float v[kC];
  const float* row = wcmp + ((size_t)h * kT + t) * kC;
  for (int c = 0; c < kC; ++c) v[c] = row[c];
  v[t >> 6] = 1.0f;                               // local block override
  unsigned mask = 0;
  for (int i = 0; i < 8; ++i) {                   // ties -> lowest index (matches top_k)
    int best = 0; float bv = v[0];
    for (int c = 1; c < kC; ++c)
      if (v[c] > bv) { bv = v[c]; best = c; }
    mask |= 1u << best;
    v[best] = -1.f;
  }
  selmask[(size_t)h * kT + t] = mask;
}

// ---------------- fused selection + sliding-window attention ----------------
template <typename MaskF>
__device__ __forceinline__ void branch_step(const f32x8* S, MaskF&& maskf,
                                            float* m, float* l, f32x8* O,
                                            const __bf16* vhead, int j0, int lane,
                                            __bf16* Plds) {
  const int half = lane >> 4, ln = lane & 15, kOff = half << 3;
  const float NEGINF = -__builtin_inff();
  float x[2][8], tm[8];
  for (int r = 0; r < 8; ++r) tm[r] = NEGINF;
  for (int nt = 0; nt < 2; ++nt)
    for (int r = 0; r < 8; ++r) {
      x[nt][r] = maskf(nt, r) ? S[nt][r] : NEGINF;
      tm[r] = fmaxf(tm[r], x[nt][r]);
    }
  for (int msk = 1; msk <= 8; msk <<= 1)
    for (int r = 0; r < 8; ++r) tm[r] = fmaxf(tm[r], __shfl_xor(tm[r], msk, 32));
  float sc[8], rs[8];
  for (int r = 0; r < 8; ++r) {
    float mn = fmaxf(m[r], tm[r]);
    sc[r] = (mn == NEGINF) ? 1.f : __expf(m[r] - mn);
    m[r] = mn;
    rs[r] = 0.f;
  }
  for (int nt = 0; nt < 2; ++nt)
    for (int r = 0; r < 8; ++r) {
      float p = (x[nt][r] == NEGINF) ? 0.f : __expf(x[nt][r] - m[r]);
      rs[r] += p;
      Plds[(r + 8 * half) * LSTR + nt * 16 + ln] = (__bf16)p;
    }
  for (int msk = 1; msk <= 8; msk <<= 1)
    for (int r = 0; r < 8; ++r) rs[r] += __shfl_xor(rs[r], msk, 32);
  for (int r = 0; r < 8; ++r) l[r] = l[r] * sc[r] + rs[r];
  for (int n = 0; n < 4; ++n)
    for (int r = 0; r < 8; ++r) O[n][r] *= sc[r];
  __syncthreads();
  bf16x16 ap = ld_frag(&Plds[ln * LSTR + kOff], &Plds[ln * LSTR + kOff + 16]);
  for (int n = 0; n < 4; ++n) {
    const __bf16* p = &vhead[(size_t)(n * 16 + ln) * kT + j0 + half * 8];
    O[n] = wmma_bf16(ap, ld_frag(p, p + 16), O[n]);
  }
  __syncthreads();
}

__global__ __launch_bounds__(32) void nsa_attn_kernel(
    const __bf16* __restrict__ qb, const __bf16* __restrict__ kb,
    const __bf16* __restrict__ vT, const unsigned* __restrict__ selmask,
    const float* __restrict__ G, const float* __restrict__ ocmp,
    __bf16* __restrict__ attn) {
  __shared__ __align__(16) __bf16 P[16 * LSTR];
  const int h = blockIdx.x, t0 = blockIdx.y * 16;
  const int lane = threadIdx.x, half = lane >> 4, ln = lane & 15, kOff = half << 3;

  const __bf16* qh = qb + (size_t)h * kT * kD;
  const __bf16* kh = kb + (size_t)h * kT * kD;
  const __bf16* vh = vT + (size_t)h * kD * kT;

  bf16x16 aq[2];
  for (int kc = 0; kc < 2; ++kc) {
    const __bf16* p = &qh[(size_t)(t0 + ln) * kD + kc * 32];
    aq[kc] = ld_frag(p + kOff, p + kOff + 16);
  }
  unsigned selm[8];
  unsigned un = 0;
  for (int r = 0; r < 8; ++r) {
    selm[r] = selmask[(size_t)h * kT + t0 + 8 * half + r];
    un |= selm[r];
  }
  for (int msk = 1; msk <= 16; msk <<= 1) un |= (unsigned)__shfl_xor((int)un, msk, 32);

  f32x8 Os[4], Ow[4];
  float ms[8], ls[8], mw[8], lw[8];
  const float NEGINF = -__builtin_inff();
  for (int n = 0; n < 4; ++n) { Os[n] = (f32x8)(0.0f); Ow[n] = (f32x8)(0.0f); }
  for (int r = 0; r < 8; ++r) { ms[r] = NEGINF; ls[r] = 0.f; mw[r] = NEGINF; lw[r] = 0.f; }

  const int bmax = (t0 + 15) >> 6;
  for (int b = 0; b <= bmax; ++b) {
    const bool want_slc = (un >> b) & 1;
    const bool want_swa_blk = (b * 64 + 63) >= (t0 - (kW - 1));
    if (!want_slc && !want_swa_blk) continue;
    for (int hb = 0; hb < 2; ++hb) {
      const int j0 = b * 64 + hb * 32;
      if (j0 > t0 + 15) break;
      // ---- scores S = q . k^T for 32 keys ----
      f32x8 S[2];
      S[0] = (f32x8)(0.0f); S[1] = (f32x8)(0.0f);
      for (int nt = 0; nt < 2; ++nt)
        for (int kc = 0; kc < 2; ++kc) {
          const __bf16* p = &kh[(size_t)(j0 + nt * 16 + ln) * kD + kc * 32 + half * 8];
          S[nt] = wmma_bf16(aq[kc], ld_frag(p, p + 16), S[nt]);
        }
      S[0] *= SCALEQ;
      S[1] *= SCALEQ;
      // ---- selection branch ----
      if (want_slc) {
        branch_step(S,
            [&](int nt, int r) -> bool {
              int key = j0 + nt * 16 + ln;
              int t = t0 + r + 8 * half;
              return (key <= t) && ((selm[r] >> b) & 1u);
            },
            ms, ls, Os, vh, j0, lane, P);
      }
      // ---- sliding-window branch ----
      if (want_swa_blk && (j0 + 31) >= (t0 - (kW - 1))) {
        branch_step(S,
            [&](int nt, int r) -> bool {
              int key = j0 + nt * 16 + ln;
              int t = t0 + r + 8 * half;
              return (key <= t) && ((t - key) < kW);
            },
            mw, lw, Ow, vh, j0, lane, P);
      }
    }
  }
  // ---- gated combine epilogue -> attn bf16 [T][H*D] ----
  float gc[8], gs[8], gw[8];
  for (int r = 0; r < 8; ++r) {
    int t = t0 + r + 8 * half;
    gc[r] = 1.f / (1.f + __expf(-G[(size_t)t * 64 + h * 3 + 0]));
    gs[r] = 1.f / (1.f + __expf(-G[(size_t)t * 64 + h * 3 + 1]));
    gw[r] = 1.f / (1.f + __expf(-G[(size_t)t * 64 + h * 3 + 2]));
  }
  for (int n = 0; n < 4; ++n)
    for (int r = 0; r < 8; ++r) {
      int t = t0 + r + 8 * half;
      int d = n * 16 + ln;
      float oslc = Os[n][r] / fmaxf(ls[r], 1e-20f);
      float oswa = Ow[n][r] / fmaxf(lw[r], 1e-20f);
      float oc = ocmp[((size_t)h * kT + t) * kD + d];
      attn[(size_t)t * kDim + h * kD + d] =
          (__bf16)(gc[r] * oc + gs[r] * oslc + gw[r] * oswa);
    }
}

// ---------------- host side ----------------
static inline size_t al256(size_t x) { return (x + 255) & ~(size_t)255; }

extern "C" void kernel_launch(void* const* d_in, const int* in_sizes, int n_in,
                              void* d_out, int out_size, void* d_ws, size_t ws_size,
                              hipStream_t stream) {
  (void)in_sizes; (void)n_in; (void)out_size; (void)ws_size;
  const float* x        = (const float*)d_in[0];
  const float* rope_cos = (const float*)d_in[1];
  const float* rope_sin = (const float*)d_in[2];
  const float* Wq       = (const float*)d_in[3];
  const float* Wk       = (const float*)d_in[4];
  const float* Wv       = (const float*)d_in[5];
  const float* Wo       = (const float*)d_in[6];
  const float* Wg       = (const float*)d_in[7];
  float* out = (float*)d_out;

  char* w = (char*)d_ws;
  size_t off = 0;
  auto alloc = [&](size_t bytes) {
    void* p = w + off;
    off = al256(off + bytes);
    return p;
  };
  __bf16*  xb     = (__bf16*)alloc((size_t)kT * kDim * 2);
  __bf16*  WqkvT  = (__bf16*)alloc((size_t)kNQKV * kDim * 2);
  __bf16*  WgT    = (__bf16*)alloc((size_t)64 * kDim * 2);
  __bf16*  WoT    = (__bf16*)alloc((size_t)kDim * kDim * 2);
  float*   qkvraw = (float*)alloc((size_t)kT * kNQKV * 4);
  float*   G      = (float*)alloc((size_t)kT * 64 * 4);
  __bf16*  qb     = (__bf16*)alloc((size_t)kH * kT * kD * 2);
  __bf16*  kb     = (__bf16*)alloc((size_t)kH * kT * kD * 2);
  __bf16*  vT     = (__bf16*)alloc((size_t)kH * kD * kT * 2);
  __bf16*  kcmp   = (__bf16*)alloc((size_t)kH * kC * kD * 2);
  __bf16*  vcmpT  = (__bf16*)alloc((size_t)kH * kD * kC * 2);
  float*   wcmp   = (float*)alloc((size_t)kH * kT * kC * 4);
  float*   ocmp   = (float*)alloc((size_t)kH * kT * kD * 4);
  unsigned* selm  = (unsigned*)alloc((size_t)kH * kT * 4);
  __bf16*  attn   = (__bf16*)alloc((size_t)kT * kDim * 2);

  // 1) pack bf16 operands
  pack_x_kernel<<<(kT * kDim) / 256, 256, 0, stream>>>(x, xb, kT * kDim);
  pack_wqkv_kernel<<<(kNQKV * kDim) / 256, 256, 0, stream>>>(Wq, Wk, Wv, WqkvT);
  pack_wg_kernel<<<(64 * kDim) / 256, 256, 0, stream>>>(Wg, WgT);
  pack_wo_kernel<<<(kDim * kDim) / 256, 256, 0, stream>>>(Wo, WoT);

  // 2) fused QKV projection (N = 3072) and gates (N = 64)
  gemm_bf16_kernel<<<dim3(kNQKV / 64, kT / 128), 256, 0, stream>>>(
      xb, WqkvT, qkvraw, kT, kNQKV, kDim);
  gemm_bf16_kernel<<<dim3(1, kT / 128), 256, 0, stream>>>(
      xb, WgT, G, kT, 64, kDim);

  // 3) RoPE + layout packing
  rope_pack_kernel<<<(kT * kDim) / 256, 256, 0, stream>>>(
      qkvraw, rope_cos, rope_sin, qb, kb, vT);

  // 4) compression branch
  pool_cmp_kernel<<<kH * kC, 64, 0, stream>>>(kb, vT, kcmp, vcmpT);
  cmp_attn_kernel<<<dim3(kH, kT / 16), 32, 0, stream>>>(qb, kcmp, vcmpT, wcmp, ocmp);
  select_topk_kernel<<<(kH * kT) / 256, 256, 0, stream>>>(wcmp, selm);

  // 5) fused selection + sliding-window attention with gated combine
  nsa_attn_kernel<<<dim3(kH, kT / 16), 32, 0, stream>>>(
      qb, kb, vT, selm, G, ocmp, attn);

  // 6) output projection
  gemm_bf16_kernel<<<dim3(kDim / 64, kT / 128), 256, 0, stream>>>(
      attn, WoT, out, kT, kDim, kDim);
}
